// LightweightNeuronLayer_75282186764739
// MI455X (gfx1250) — matrix operand order repaired and checked by reading, not compiled, not run on hardware
//
#include <hip/hip_runtime.h>
#include <hip/hip_bf16.h>

typedef __attribute__((ext_vector_type(16))) __bf16 v16bf;
typedef __attribute__((ext_vector_type(8)))  __bf16 v8bf;
typedef __attribute__((ext_vector_type(8)))  float  v8f;
typedef __attribute__((ext_vector_type(4)))  float  v4f;

#define S_IN   4096   // INPUT_SIZE
#define N_OUT  4096   // NUM_NEURONS
#define ATTN   64
#define BATCH  8192

__device__ __forceinline__ unsigned short f2bf(float f) {
  unsigned u = __float_as_uint(f);
  u += 0x7FFFu + ((u >> 16) & 1u);   // round-to-nearest-even
  return (unsigned short)(u >> 16);
}

// ---------------------------------------------------------------------------
// Kernel 1: Wn f32 -> bf16 (row-major [N_OUT, S_IN])
// ---------------------------------------------------------------------------
__global__ void __launch_bounds__(256) cvt_wn(const float* __restrict__ Wn,
                                              unsigned short* __restrict__ W16) {
  int idx = (blockIdx.x * 256 + threadIdx.x) * 4;
  v4f w = *reinterpret_cast<const v4f*>(Wn + idx);
  ushort4 o;
  o.x = f2bf(w.x); o.y = f2bf(w.y); o.z = f2bf(w.z); o.w = f2bf(w.w);
  *reinterpret_cast<ushort4*>(W16 + idx) = o;
}

// ---------------------------------------------------------------------------
// Kernel 2: h = x + bo + Wo @ (Wv @ x_b + bv), store bf16 [BATCH, S_IN]
// One 256-thread block (8 waves) per batch row. Wave w owns v-outputs
// a in [8w, 8w+8): 8 accumulators/thread, 40 shuffles, no LDS reduction.
// ---------------------------------------------------------------------------
__global__ void __launch_bounds__(256) build_h(const float* __restrict__ x,
                                               const float* __restrict__ Wv,
                                               const float* __restrict__ bv,
                                               const float* __restrict__ Wo,
                                               const float* __restrict__ bo,
                                               unsigned short* __restrict__ H16) {
  __shared__ float vsh[ATTN];        // final v for this row

  const int b    = blockIdx.x;
  const int tid  = threadIdx.x;
  const int lane = tid & 31;
  const int wave = tid >> 5;
  const float* __restrict__ xr = x + (size_t)b * S_IN;
  const float* __restrict__ wvr = Wv + (size_t)(wave * 8) * S_IN;

  // v[8w+j] = sum_k x[b][k] * Wv[8w+j][k]
  float acc[8];
#pragma unroll
  for (int j = 0; j < 8; ++j) acc[j] = 0.f;

  for (int i = 0; i < S_IN / 32; ++i) {
    int k = lane + i * 32;
    float xv = xr[k];
#pragma unroll
    for (int j = 0; j < 8; ++j)
      acc[j] += xv * wvr[j * S_IN + k];
  }

#pragma unroll
  for (int j = 0; j < 8; ++j) {
    float s = acc[j];
    s += __shfl_down(s, 16);
    s += __shfl_down(s, 8);
    s += __shfl_down(s, 4);
    s += __shfl_down(s, 2);
    s += __shfl_down(s, 1);
    if (lane == 0) vsh[wave * 8 + j] = s + bv[wave * 8 + j];
  }
  __syncthreads();

  // h[s] = x[b][s] + bo[s] + dot(Wo[s][:], v)
  for (int j = 0; j < S_IN / 256; ++j) {
    int s = tid + j * 256;
    float hv = xr[s] + bo[s];
    const v4f* __restrict__ wo = reinterpret_cast<const v4f*>(Wo + (size_t)s * ATTN);
#pragma unroll
    for (int a4 = 0; a4 < ATTN / 4; ++a4) {
      v4f w = wo[a4];
      hv += vsh[a4 * 4 + 0] * w.x + vsh[a4 * 4 + 1] * w.y +
            vsh[a4 * 4 + 2] * w.z + vsh[a4 * 4 + 3] * w.w;
    }
    H16[(size_t)b * S_IN + s] = f2bf(hv);
  }
}

// ---------------------------------------------------------------------------
// Kernel 3: out = relu(H @ W^T + bn), H [BATCH,S_IN] bf16, W [N_OUT,S_IN] bf16
// Block tile 128(M) x 256(N), 8 waves in 2(M) x 4(N) grid; each wave computes
// a 64x64 tile via 4x4 accumulators of v_wmma_f32_16x16x32_bf16.
// Per k-step/wave: 16 global_load_b128 -> 16 wmma (1.0 loads per wmma).
// ---------------------------------------------------------------------------
__global__ void __launch_bounds__(256) gemm_relu(const unsigned short* __restrict__ H,
                                                 const unsigned short* __restrict__ W,
                                                 const float* __restrict__ bnb,
                                                 float* __restrict__ out) {
  const int lane = threadIdx.x & 31;
  const int wave = threadIdx.x >> 5;
  const int wm   = wave >> 2;      // 0..1 : wave row (64 M-rows each)
  const int wn   = wave & 3;       // 0..3 : wave col (64 N-cols each)
  const int r    = lane & 15;
  const int hi   = lane >> 4;      // lane half selects K sub-chunk
  const int bm   = blockIdx.y * 128;
  const int bn0  = blockIdx.x * 256;

  // A fragment (16x32 bf16, M x K): lane r holds row bm+...+r.
  //   elems 0..7  = H[row][k0 + hi*8  + 0..7]
  //   elems 8..15 = H[row][k0 + 16 + hi*8 + 0..7]
  const unsigned short* __restrict__ Abase =
      H + (size_t)(bm + wm * 64 + r) * S_IN + hi * 8;
  // B fragment (32x16 bf16, K x N): lane r holds column n = row of W.
  //   elems 0..7  = W[col][k0 + hi*8  + 0..7]
  //   elems 8..15 = W[col][k0 + 16 + hi*8 + 0..7]
  const unsigned short* __restrict__ Bbase =
      W + (size_t)(bn0 + wn * 64 + r) * S_IN + hi * 8;

  v8f zero;
#pragma unroll
  for (int e = 0; e < 8; ++e) zero[e] = 0.f;
  v8f acc[4][4];
#pragma unroll
  for (int mt = 0; mt < 4; ++mt)
#pragma unroll
    for (int nt = 0; nt < 4; ++nt) acc[mt][nt] = zero;

  for (int k0 = 0; k0 < S_IN; k0 += 32) {
    v16bf av[4];
#pragma unroll
    for (int mt = 0; mt < 4; ++mt) {
      v8bf lo  = *reinterpret_cast<const v8bf*>(Abase + (size_t)mt * 16 * S_IN + k0);
      v8bf hi8 = *reinterpret_cast<const v8bf*>(Abase + (size_t)mt * 16 * S_IN + k0 + 16);
#pragma unroll
      for (int e = 0; e < 8; ++e) { av[mt][e] = lo[e]; av[mt][e + 8] = hi8[e]; }
    }
    v16bf bvv[4];
#pragma unroll
    for (int nt = 0; nt < 4; ++nt) {
      v8bf lo  = *reinterpret_cast<const v8bf*>(Bbase + (size_t)nt * 16 * S_IN + k0);
      v8bf hi8 = *reinterpret_cast<const v8bf*>(Bbase + (size_t)nt * 16 * S_IN + k0 + 16);
#pragma unroll
      for (int e = 0; e < 8; ++e) { bvv[nt][e] = lo[e]; bvv[nt][e + 8] = hi8[e]; }
    }
#pragma unroll
    for (int mt = 0; mt < 4; ++mt)
#pragma unroll
      for (int nt = 0; nt < 4; ++nt)
        acc[mt][nt] = __builtin_amdgcn_wmma_f32_16x16x32_bf16(
            false, av[mt], false, bvv[nt], (short)0, acc[mt][nt], false, false);
  }

  // Epilogue: C/D layout — lane (hi,r): elem i -> row m0+hi*8+i, col n0+r
#pragma unroll
  for (int nt = 0; nt < 4; ++nt) {
    int col = bn0 + wn * 64 + nt * 16 + r;
    float bias = bnb[col];
#pragma unroll
    for (int mt = 0; mt < 4; ++mt) {
#pragma unroll
      for (int e = 0; e < 8; ++e) {
        int row = bm + wm * 64 + mt * 16 + hi * 8 + e;
        float val = acc[mt][nt][e] + bias;
        out[(size_t)row * N_OUT + col] = val > 0.f ? val : 0.f;
      }
    }
  }
}

// ---------------------------------------------------------------------------
// Host launcher
// d_in order: x, Wq, bq, Wk, bk, Wv, bv, Wo, bo, Wn, bn
// q/k branch is dead (softmax over one score == 1).
// ---------------------------------------------------------------------------
extern "C" void kernel_launch(void* const* d_in, const int* in_sizes, int n_in,
                              void* d_out, int out_size, void* d_ws, size_t ws_size,
                              hipStream_t stream) {
  const float* x   = (const float*)d_in[0];
  const float* Wv  = (const float*)d_in[5];
  const float* bv  = (const float*)d_in[6];
  const float* Wo  = (const float*)d_in[7];
  const float* bo  = (const float*)d_in[8];
  const float* Wn  = (const float*)d_in[9];
  const float* bnb = (const float*)d_in[10];

  unsigned short* H16 = (unsigned short*)d_ws;                                     // 64 MiB
  unsigned short* W16 = (unsigned short*)((char*)d_ws + (size_t)BATCH * S_IN * 2); // +32 MiB

  cvt_wn<<<(N_OUT * S_IN) / 1024, 256, 0, stream>>>(Wn, W16);
  build_h<<<BATCH, 256, 0, stream>>>(x, Wv, bv, Wo, bo, H16);

  dim3 grid(N_OUT / 256, BATCH / 128);
  gemm_relu<<<grid, 256, 0, stream>>>(H16, W16, bnb, (float*)d_out);
}
